// Item_GraphConvolution_reconstruct_57698590654941
// MI455X (gfx1250) — matrix operand order repaired and checked by reading, not compiled, not run on hardware
//
#include <hip/hip_runtime.h>
#include <hip/hip_bf16.h>
#include <math.h>

// ---------------------------------------------------------------------------
// Types for CDNA5 WMMA (wave32): v_wmma_f32_16x16x32_bf16
// ---------------------------------------------------------------------------
typedef __attribute__((ext_vector_type(16))) __bf16        v16bf;
typedef __attribute__((ext_vector_type(8)))  float         v8f;
typedef __attribute__((ext_vector_type(4)))  unsigned int  v4u;
typedef __attribute__((ext_vector_type(8)))  unsigned int  v8u;
typedef __attribute__((ext_vector_type(4)))  float         v4f;

#define N_NODES 4096
#define EMB     128

__device__ __forceinline__ unsigned short f2bf(float f) {
  unsigned int u = __builtin_bit_cast(unsigned int, f);
  u += 0x7FFFu + ((u >> 16) & 1u);   // round-to-nearest-even
  return (unsigned short)(u >> 16);
}
__device__ __forceinline__ float bf2f(unsigned short s) {
  return __builtin_bit_cast(float, (unsigned int)s << 16);
}
// pack 8 f32 -> 8 bf16 in a uint4
__device__ __forceinline__ v4u pack8(v4f a, v4f b) {
  v4u r;
  r[0] = (unsigned)f2bf(a[0]) | ((unsigned)f2bf(a[1]) << 16);
  r[1] = (unsigned)f2bf(a[2]) | ((unsigned)f2bf(a[3]) << 16);
  r[2] = (unsigned)f2bf(b[0]) | ((unsigned)f2bf(b[1]) << 16);
  r[3] = (unsigned)f2bf(b[2]) | ((unsigned)f2bf(b[3]) << 16);
  return r;
}

// Load a 16-element bf16 fragment as two 16B LDS loads (ds_load_b128).
// off1 = 16 for both A- and B-fragments.
__device__ __forceinline__ v16bf ld_frag(const unsigned short* p, int off1) {
  v4u lo = *(const v4u*)(p);
  v4u hi = *(const v4u*)(p + off1);
  v8u c  = __builtin_shufflevector(lo, hi, 0, 1, 2, 3, 4, 5, 6, 7);
  return __builtin_bit_cast(v16bf, c);
}

__device__ __forceinline__ v8f wmma_bf16(v16bf a, v16bf b, v8f c) {
  return __builtin_amdgcn_wmma_f32_16x16x32_bf16(
      false, a, false, b, (short)0, c, false, false);
}

// ---------------------------------------------------------------------------
// GEMM: C[M,128] = A[M,K] @ B[K,128]; B given in BT layout: BT[n*ldbT + k]
// (bf16). Block = 256 threads (8 waves), tile 128(M) x 128(N), BK = 64.
// Each wave owns a 16x128 strip -> 16 v_wmma per k-tile.
//  A_MODE: 0 = f32 row-major (convert on load), 1 = bf16 row-major,
//          2 = concat of two bf16 [M,128] sources along K (K = 256)
//  EP    : 0 = none, 1 = relu, 2 = +bias1[n]+bias2[n]
// Outputs: outF (f32 row-major, optional), outT (bf16 BT layout [128][M],
//          stride 4096, optional -> feeds the next GEMM's B operand).
// ---------------------------------------------------------------------------
#define BK 64
#define PA 72   // LDS pitch (ushorts): 64 + 8 pad
#define PB 72

template <int A_MODE, int EP>
__global__ __launch_bounds__(256) void gemm_bf16_n128(
    const float* __restrict__ Af, const unsigned short* __restrict__ Abf,
    const unsigned short* __restrict__ Abf2, int K, int lda,
    const unsigned short* __restrict__ BT, int ldbT,
    const float* __restrict__ bias1, const float* __restrict__ bias2,
    float* __restrict__ outF, unsigned short* __restrict__ outT)
{
  __shared__ __align__(16) unsigned short ldsA[128 * PA];
  __shared__ __align__(16) unsigned short ldsB[128 * PB];

  const int m0   = blockIdx.x * 128;
  const int wave = threadIdx.x >> 5;
  const int lane = threadIdx.x & 31;
  const int hi   = lane >> 4;
  const int lr   = lane & 15;

  v8f acc[8];
#pragma unroll
  for (int i = 0; i < 8; ++i) acc[i] = (v8f)(0.0f);

  for (int k0 = 0; k0 < K; k0 += BK) {
    // ---- stage A tile: 128 rows x 64 K, 4 x uint4 per thread ----
#pragma unroll
    for (int i = 0; i < 4; ++i) {
      int c   = threadIdx.x + i * 256;   // chunk of 8 bf16
      int row = c >> 3;                  // 8 chunks per row
      int kq  = (c & 7) * 8;
      v4u val;
      if (A_MODE == 0) {
        const float* src = Af + (size_t)(m0 + row) * lda + k0 + kq;
        v4f f0 = *(const v4f*)(src);
        v4f f1 = *(const v4f*)(src + 4);
        val = pack8(f0, f1);
      } else if (A_MODE == 1) {
        val = *(const v4u*)(Abf + (size_t)(m0 + row) * lda + k0 + kq);
      } else {
        const unsigned short* base =
            (k0 < 128) ? (Abf + k0) : (Abf2 + (k0 - 128));
        val = *(const v4u*)(base + (size_t)(m0 + row) * 128 + kq);
      }
      *(v4u*)(&ldsA[row * PA + kq]) = val;
    }
    // ---- stage B tile from BT layout: 128 cols x 64 K, vectorized ----
#pragma unroll
    for (int i = 0; i < 4; ++i) {
      int c  = threadIdx.x + i * 256;
      int n  = c >> 3;
      int kq = (c & 7) * 8;
      v4u val = *(const v4u*)(BT + (size_t)n * ldbT + k0 + kq);
      *(v4u*)(&ldsB[n * PB + kq]) = val;
    }
    __syncthreads();

#pragma unroll
    for (int ks = 0; ks < BK; ks += 32) {
      v16bf afrag = ld_frag(&ldsA[(wave * 16 + lr) * PA + ks + hi * 8], 16);
#pragma unroll
      for (int nt = 0; nt < 8; ++nt) {
        v16bf bfrag = ld_frag(&ldsB[(nt * 16 + lr) * PB + ks + hi * 8], 16);
        acc[nt] = wmma_bf16(afrag, bfrag, acc[nt]);
      }
    }
    __syncthreads();
  }

  // Epilogue. D layout: VGPR v -> M = v + 8*hi, N = lr (per 16x16 tile).
#pragma unroll
  for (int nt = 0; nt < 8; ++nt) {
    int n = nt * 16 + lr;
    float badd = (EP == 2) ? (bias1[n] + bias2[n]) : 0.0f;
#pragma unroll
    for (int v = 0; v < 8; ++v) {
      int r = m0 + wave * 16 + hi * 8 + v;
      float x = acc[nt][v] + badd;
      if (EP == 1) x = fmaxf(x, 0.0f);
      if (outF) outF[(size_t)r * 128 + n] = x;
      if (outT) outT[(size_t)n * N_NODES + r] = f2bf(x);  // BT layout
    }
  }
}

// ---------------------------------------------------------------------------
// f32 -> bf16 converters
// ---------------------------------------------------------------------------
__global__ void conv_bf16_vec(const float* __restrict__ src,
                              unsigned short* __restrict__ dst, int nchunks)
{
  int c = blockIdx.x * blockDim.x + threadIdx.x;   // chunk of 8
  if (c < nchunks) {
    v4f f0 = *(const v4f*)(src + (size_t)c * 8);
    v4f f1 = *(const v4f*)(src + (size_t)c * 8 + 4);
    *(v4u*)(dst + (size_t)c * 8) = pack8(f0, f1);
  }
}

// wT[n*K + k] = f2bf(w[k*N + n])   (weight [K,N] -> BT layout)
__global__ void conv_bf16_transpose(const float* __restrict__ w,
                                    unsigned short* __restrict__ wT,
                                    int K, int N)
{
  int idx = blockIdx.x * blockDim.x + threadIdx.x;
  if (idx < K * N) {
    int n = idx / K, k = idx - n * K;
    wT[(size_t)n * K + k] = f2bf(w[(size_t)k * N + n]);
  }
}

// low = S1 + support ; mid = S2 - support  (bf16, vectorized by 4)
__global__ void make_low_mid(const float* __restrict__ S1,
                             const float* __restrict__ S2,
                             const float* __restrict__ sup,
                             unsigned short* __restrict__ lowbf,
                             unsigned short* __restrict__ midbf, int nquads)
{
  int q = blockIdx.x * blockDim.x + threadIdx.x;
  if (q < nquads) {
    v4f s1 = *(const v4f*)(S1 + (size_t)q * 4);
    v4f s2 = *(const v4f*)(S2 + (size_t)q * 4);
    v4f sp = *(const v4f*)(sup + (size_t)q * 4);
    unsigned int lo0 = (unsigned)f2bf(s1[0] + sp[0]) | ((unsigned)f2bf(s1[1] + sp[1]) << 16);
    unsigned int lo1 = (unsigned)f2bf(s1[2] + sp[2]) | ((unsigned)f2bf(s1[3] + sp[3]) << 16);
    unsigned int mi0 = (unsigned)f2bf(s2[0] - sp[0]) | ((unsigned)f2bf(s2[1] - sp[1]) << 16);
    unsigned int mi1 = (unsigned)f2bf(s2[2] - sp[2]) | ((unsigned)f2bf(s2[3] - sp[3]) << 16);
    ((unsigned int*)lowbf)[(size_t)q * 2]     = lo0;
    ((unsigned int*)lowbf)[(size_t)q * 2 + 1] = lo1;
    ((unsigned int*)midbf)[(size_t)q * 2]     = mi0;
    ((unsigned int*)midbf)[(size_t)q * 2 + 1] = mi1;
  }
}

// ---------------------------------------------------------------------------
// Fused reconstruction loss over 32x64 tiles. G = low.low^T / mid.mid^T via
// WMMA (K = 128 fully LDS-resident), BCE against bf16 adjacency (exact 0/1),
// block reduce, one atomicAdd per block.
// ---------------------------------------------------------------------------
#define LP 136

__device__ __forceinline__ float sigmoidf_(float x) {
  return 1.0f / (1.0f + expf(-x));
}

__global__ __launch_bounds__(256) void recon_loss_kernel(
    const unsigned short* __restrict__ lowbf,
    const unsigned short* __restrict__ midbf,
    const unsigned short* __restrict__ adjB, float* __restrict__ accum)
{
  __shared__ __align__(16) unsigned short Al[32 * LP];
  __shared__ __align__(16) unsigned short Bl[64 * LP];
  __shared__ __align__(16) unsigned short Am[32 * LP];
  __shared__ __align__(16) unsigned short Bm[64 * LP];
  __shared__ float red[256];

  const int i0 = blockIdx.y * 32;
  const int j0 = blockIdx.x * 64;

  // rows are 128 bf16 = 16 uint4; A tiles: 512 chunks, B tiles: 1024 chunks
#pragma unroll
  for (int i = 0; i < 2; ++i) {
    int c = threadIdx.x + i * 256;
    int r = c >> 4, kq = (c & 15) * 8;
    *(v4u*)(&Al[r * LP + kq]) = *(const v4u*)(lowbf + (size_t)(i0 + r) * 128 + kq);
    *(v4u*)(&Am[r * LP + kq]) = *(const v4u*)(midbf + (size_t)(i0 + r) * 128 + kq);
  }
#pragma unroll
  for (int i = 0; i < 4; ++i) {
    int c = threadIdx.x + i * 256;
    int r = c >> 4, kq = (c & 15) * 8;
    *(v4u*)(&Bl[r * LP + kq]) = *(const v4u*)(lowbf + (size_t)(j0 + r) * 128 + kq);
    *(v4u*)(&Bm[r * LP + kq]) = *(const v4u*)(midbf + (size_t)(j0 + r) * 128 + kq);
  }
  __syncthreads();

  const int wave = threadIdx.x >> 5;
  const int wm = wave >> 2, wn = wave & 3;
  const int lane = threadIdx.x & 31;
  const int hi = lane >> 4, lr = lane & 15;

  v8f accL = (v8f)(0.0f), accM = (v8f)(0.0f);
#pragma unroll
  for (int ks = 0; ks < 4; ++ks) {
    int kb = ks * 32;
    v16bf aL = ld_frag(&Al[(wm * 16 + lr) * LP + kb + hi * 8], 16);
    v16bf bL = ld_frag(&Bl[(wn * 16 + lr) * LP + kb + hi * 8], 16);
    accL = wmma_bf16(aL, bL, accL);
    v16bf aM = ld_frag(&Am[(wm * 16 + lr) * LP + kb + hi * 8], 16);
    v16bf bM = ld_frag(&Bm[(wn * 16 + lr) * LP + kb + hi * 8], 16);
    accM = wmma_bf16(aM, bM, accM);
  }

  float part = 0.0f;
#pragma unroll
  for (int v = 0; v < 8; ++v) {
    int gi = i0 + wm * 16 + hi * 8 + v;
    int gj = j0 + wn * 16 + lr;
    float rec = sigmoidf_(accL[v]) + sigmoidf_(accM[v]);
    rec = fminf(rec, 1.0f);
    float a  = bf2f(adjB[(size_t)gi * N_NODES + gj]);   // exact 0/1
    float lp = fmaxf(logf(rec),    -100.0f);
    float l1 = fmaxf(log1pf(-rec), -100.0f);
    part += a * lp + (1.0f - a) * l1;
  }

  red[threadIdx.x] = part;
  __syncthreads();
  for (int s = 128; s > 0; s >>= 1) {
    if (threadIdx.x < s) red[threadIdx.x] += red[threadIdx.x + s];
    __syncthreads();
  }
  if (threadIdx.x == 0) atomicAdd(accum, red[0]);
}

__global__ void zero_accum(float* p) { p[0] = 0.0f; }

__global__ void finalize_loss(const float* __restrict__ accum,
                              float* __restrict__ out) {
  out[0] = -accum[0] / ((float)N_NODES * (float)N_NODES);
}

// ---------------------------------------------------------------------------
extern "C" void kernel_launch(void* const* d_in, const int* in_sizes, int n_in,
                              void* d_out, int out_size, void* d_ws, size_t ws_size,
                              hipStream_t stream)
{
  (void)in_sizes; (void)n_in; (void)out_size; (void)ws_size;

  const float* feature = (const float*)d_in[0];  // [4096,512]
  const float* weight  = (const float*)d_in[1];  // [512,128]
  const float* bias    = (const float*)d_in[2];  // [128]
  const float* cat_w   = (const float*)d_in[3];  // [128,256]
  const float* cat_b   = (const float*)d_in[4];  // [128]
  const float* adj     = (const float*)d_in[5];  // [4096,4096]
  float* out = (float*)d_out;                    // [4096*128] ++ [1]

  const size_t NE = (size_t)N_NODES * EMB;       // 524288
  const size_t NN = (size_t)N_NODES * N_NODES;   // 16.7M
  char* ws = (char*)d_ws;
  float*          supF = (float*)ws;           ws += NE * 4;
  float*          S1F  = (float*)ws;           ws += NE * 4;
  float*          S2F  = (float*)ws;           ws += NE * 4;
  unsigned short* supT = (unsigned short*)ws;  ws += NE * 2;  // BT [128][4096]
  unsigned short* S1T  = (unsigned short*)ws;  ws += NE * 2;  // BT [128][4096]
  unsigned short* lowB = (unsigned short*)ws;  ws += NE * 2;
  unsigned short* midB = (unsigned short*)ws;  ws += NE * 2;
  unsigned short* adjB = (unsigned short*)ws;  ws += NN * 2;  // 32 MB
  unsigned short* wT   = (unsigned short*)ws;  ws += 512 * 128 * 2;
  unsigned short* catB = (unsigned short*)ws;  ws += 128 * 256 * 2;
  float*          acc  = (float*)ws;           ws += 16;

  const dim3 gblk(N_NODES / 128);   // 32 GEMM blocks
  const dim3 thr(256);

  zero_accum<<<1, 1, 0, stream>>>(acc);

  // 0) one-time conversions: adj -> bf16 (exact), weight -> bf16 BT,
  //    cat_w -> bf16 (already [n][k] = BT layout, ldbT = 256)
  conv_bf16_vec<<<(int)(NN / 8 / 256), thr, 0, stream>>>(adj, adjB, (int)(NN / 8));
  conv_bf16_transpose<<<(512 * 128 + 255) / 256, thr, 0, stream>>>(weight, wT, 512, 128);
  conv_bf16_vec<<<(128 * 256 / 8 + 255) / 256, thr, 0, stream>>>(cat_w, catB, 128 * 256 / 8);

  // 1) support = relu(feature @ weight) -> f32 + BT bf16
  gemm_bf16_n128<0, 1><<<gblk, thr, 0, stream>>>(
      feature, nullptr, nullptr, 512, 512, wT, 512,
      nullptr, nullptr, supF, supT);

  // 2) S1 = adj @ support
  gemm_bf16_n128<1, 0><<<gblk, thr, 0, stream>>>(
      nullptr, adjB, nullptr, N_NODES, N_NODES, supT, N_NODES,
      nullptr, nullptr, S1F, S1T);

  // 3) S2 = adj @ S1   (== adj@adj@support without the N^3 GEMM)
  gemm_bf16_n128<1, 0><<<gblk, thr, 0, stream>>>(
      nullptr, adjB, nullptr, N_NODES, N_NODES, S1T, N_NODES,
      nullptr, nullptr, S2F, nullptr);

  // 4) low = S1 + support ; mid = S2 - support  (bf16 row-major)
  make_low_mid<<<(int)(NE / 4 / 256), thr, 0, stream>>>(
      S1F, S2F, supF, lowB, midB, (int)(NE / 4));

  // 5) out = concat(low,mid) @ cat_w^T + cat_b + bias
  gemm_bf16_n128<2, 2><<<gblk, thr, 0, stream>>>(
      nullptr, lowB, midB, 256, 128, catB, 256,
      cat_b, bias, out, nullptr);

  // 6) BCE reconstruction loss (fused Gram + sigmoid + log + reduce)
  recon_loss_kernel<<<dim3(N_NODES / 64, N_NODES / 32), thr, 0, stream>>>(
      lowB, midB, adjB, acc);

  // 7) loss scalar
  finalize_loss<<<1, 1, 0, stream>>>(acc, out + NE);
}
